// GAT_18356690223923
// MI455X (gfx1250) — matrix operand order repaired and checked by reading, not compiled, not run on hardware
//
#include <hip/hip_runtime.h>
#include <hip/hip_bf16.h>
#include <math.h>

// ---------------- CDNA5 WMMA types ----------------
typedef __attribute__((ext_vector_type(16))) __bf16 v16bf;
typedef __attribute__((ext_vector_type(8)))  float  v8f;

// ---------------- helpers ----------------
__device__ __forceinline__ unsigned short f2bf(float f) {
    unsigned int u = __float_as_uint(f);
    unsigned int r = (u + 0x7FFFu + ((u >> 16) & 1u)) >> 16;   // RNE
    return (unsigned short)r;
}

__device__ __forceinline__ void atomicMaxF(float* addr, float v) {
    if (v >= 0.0f) atomicMax((int*)addr, __float_as_int(v));
    else           atomicMin((unsigned int*)addr, __float_as_uint(v));
}

// ---------------- conversion kernels ----------------
__global__ void k_f32_to_bf16(const float* __restrict__ src,
                              unsigned short* __restrict__ dst, int n) {
    int i = blockIdx.x * blockDim.x + threadIdx.x;
    if (i < n) dst[i] = f2bf(src[i]);
}

// Wt[n][k] = W[k][n], zero-padded to Npad output columns
__global__ void k_w_to_bf16_t(const float* __restrict__ W,
                              unsigned short* __restrict__ Wt,
                              int K, int Ncols, int Npad) {
    int i = blockIdx.x * blockDim.x + threadIdx.x;
    if (i >= Npad * K) return;
    int n = i / K, k = i - n * K;
    float v = (n < Ncols) ? W[k * Ncols + n] : 0.0f;
    Wt[i] = f2bf(v);
}

__global__ void k_fill(float* __restrict__ p, float v, int n) {
    int i = blockIdx.x * blockDim.x + threadIdx.x;
    if (i < n) p[i] = v;
}

// ---------------- WMMA GEMM: C[M,ldc] = A[M,K](bf16) * Wt[Npad,K]^T (bf16) ----------------
// One wave (32 threads) per 16x16 output tile. grid = (M/16, Npad/16).
__global__ void k_wmma_gemm_bf16(const unsigned short* __restrict__ A,
                                 const unsigned short* __restrict__ Bt,
                                 float* __restrict__ C,
                                 int M, int K, int Nout, int ldc) {
    const int tm   = blockIdx.x;
    const int tn   = blockIdx.y;
    const int lane = threadIdx.x & 31;
    const int hi   = lane >> 4;        // 0 or 1
    const int l15  = lane & 15;

    const int m   = tm * 16 + l15;     // A row for this lane (both halves)
    const int n   = tn * 16 + l15;     // B column for this lane
    const int kbA = hi * 8;            // A-frag K base per half
    const int kbB = hi * 8;            // B-frag K base per half

    const unsigned short* pa = A  + (long long)m * K + kbA;
    const unsigned short* pb = Bt + (long long)n * K + kbB;

    v8f acc = {};
    for (int k = 0; k < K; k += 32) {
        union { v16bf v; uint4 q[2]; } fa, fb;
        fa.q[0] = *reinterpret_cast<const uint4*>(pa + k);        // K = kbA + 0..7
        fa.q[1] = *reinterpret_cast<const uint4*>(pa + k + 16);   // K = kbA + 16..23
        fb.q[0] = *reinterpret_cast<const uint4*>(pb + k);        // K = kbB + 0..7
        fb.q[1] = *reinterpret_cast<const uint4*>(pb + k + 16);   // K = kbB + 16..23
        acc = __builtin_amdgcn_wmma_f32_16x16x32_bf16(
                  false, fa.v, false, fb.v, (short)0, acc, false, false);
    }

#pragma unroll
    for (int r = 0; r < 8; ++r) {
        int row = tm * 16 + r + hi * 8;
        int col = tn * 16 + l15;
        if (row < M && col < Nout)
            C[(long long)row * ldc + col] = acc[r];
    }
}

// ---------------- attention scores: al[n,h] = sum_c xl[n,h,c]*a[h,c] ----------------
__global__ void k_scores(const float* __restrict__ xl,
                         const float* __restrict__ asrc,
                         const float* __restrict__ adst,
                         float* __restrict__ alsrc, float* __restrict__ aldst,
                         int N, int H, int C) {
    int i = blockIdx.x * blockDim.x + threadIdx.x;
    if (i >= N * H) return;
    int n = i / H, h = i - n * H;
    const float* v = xl + (long long)n * H * C + h * C;
    float s0 = 0.f, s1 = 0.f;
    for (int c = 0; c < C; ++c) {
        float x = v[c];
        s0 += x * asrc[h * C + c];
        s1 += x * adst[h * C + c];
    }
    alsrc[i] = s0;
    aldst[i] = s1;
}

// ---------------- edge pass 1: segment max of leaky_relu(alpha) over dst ----------------
__global__ void k_edge_max(const int* __restrict__ src, const int* __restrict__ dst,
                           const float* __restrict__ alsrc, const float* __restrict__ aldst,
                           float* __restrict__ amax, int E, int H) {
    int i = blockIdx.x * blockDim.x + threadIdx.x;
    if (i >= E * H) return;
    int e = i / H, h = i - e * H;
    int s = src[e], d = dst[e];
    float a = alsrc[s * H + h] + aldst[d * H + h];
    a = (a > 0.f) ? a : 0.2f * a;
    atomicMaxF(&amax[d * H + h], a);
}

__global__ void k_fixmax(float* __restrict__ amax, int n) {
    int i = blockIdx.x * blockDim.x + threadIdx.x;
    if (i >= n) return;
    float v = amax[i];
    if (!(v > -3.0e38f)) amax[i] = 0.0f;   // -inf -> 0 (matches isfinite replacement)
}

// ---------------- edge pass 2: denom[d,h] += exp(alpha - amax[d,h]) ----------------
__global__ void k_edge_sum(const int* __restrict__ src, const int* __restrict__ dst,
                           const float* __restrict__ alsrc, const float* __restrict__ aldst,
                           const float* __restrict__ amax, float* __restrict__ denom,
                           int E, int H) {
    int i = blockIdx.x * blockDim.x + threadIdx.x;
    if (i >= E * H) return;
    int e = i / H, h = i - e * H;
    int s = src[e], d = dst[e];
    float a = alsrc[s * H + h] + aldst[d * H + h];
    a = (a > 0.f) ? a : 0.2f * a;
    atomicAdd(&denom[d * H + h], expf(a - amax[d * H + h]));
}

// ---------------- edge pass 3: out[d,f] += xl[s,f] * coef(e,h) ----------------
// one thread per (edge, feature); consecutive threads -> consecutive features (coalesced)
__global__ void k_edge_scatter(const int* __restrict__ src, const int* __restrict__ dst,
                               const float* __restrict__ alsrc, const float* __restrict__ aldst,
                               const float* __restrict__ amax, const float* __restrict__ denom,
                               const float* __restrict__ xl, float* __restrict__ out,
                               int E, int H, int C) {
    long long i = (long long)blockIdx.x * blockDim.x + threadIdx.x;
    int HC = H * C;
    long long total = (long long)E * HC;
    if (i >= total) return;
    int e = (int)(i / HC);
    int f = (int)(i - (long long)e * HC);
    int h = f / C;
    int s = src[e], d = dst[e];
    float a = alsrc[s * H + h] + aldst[d * H + h];
    a = (a > 0.f) ? a : 0.2f * a;
    float w    = expf(a - amax[d * H + h]);
    float coef = w / (denom[d * H + h] + 1e-16f);
    atomicAdd(&out[(long long)d * HC + f], xl[(long long)s * HC + f] * coef);
}

// ---------------- BN stats (biased var), one block per feature column ----------------
__global__ void k_bn_stats(const float* __restrict__ x, const float* __restrict__ bias,
                           float* __restrict__ stats, int N, int F) {
    int f = blockIdx.x;
    float b = bias[f];
    float s = 0.f, s2 = 0.f;
    for (int n = threadIdx.x; n < N; n += 256) {
        float v = x[(long long)n * F + f] + b;
        s += v; s2 += v * v;
    }
    __shared__ float sh[256], sh2[256];
    sh[threadIdx.x] = s; sh2[threadIdx.x] = s2;
    __syncthreads();
    for (int o = 128; o > 0; o >>= 1) {
        if ((int)threadIdx.x < o) {
            sh[threadIdx.x]  += sh[threadIdx.x + o];
            sh2[threadIdx.x] += sh2[threadIdx.x + o];
        }
        __syncthreads();
    }
    if (threadIdx.x == 0) {
        float mu = sh[0] / (float)N;
        stats[f]     = mu;
        stats[F + f] = sh2[0] / (float)N - mu * mu;
    }
}

// ---------------- (x + bias - mu)/sqrt(var+eps)*g + beta, then ELU ----------------
__global__ void k_bn_elu(const float* __restrict__ x, const float* __restrict__ bias,
                         const float* __restrict__ stats, const float* __restrict__ g,
                         const float* __restrict__ beta, float* __restrict__ y,
                         int N, int F) {
    long long i = (long long)blockIdx.x * blockDim.x + threadIdx.x;
    if (i >= (long long)N * F) return;
    int f = (int)(i % F);
    float v   = x[i] + bias[f];
    float mu  = stats[f];
    float var = stats[F + f];
    float z = (v - mu) * rsqrtf(var + 1e-5f) * g[f] + beta[f];
    y[i] = (z > 0.f) ? z : (expf(z) - 1.f);
}

// ---------------- final: log_softmax(x + bias) per row ----------------
__global__ void k_logsoftmax(const float* __restrict__ x, const float* __restrict__ bias,
                             float* __restrict__ out, int N, int C) {
    int n = blockIdx.x * blockDim.x + threadIdx.x;
    if (n >= N) return;
    float v[16];
    float mx = -3.4e38f;
    for (int c = 0; c < C; ++c) {
        v[c] = x[(long long)n * C + c] + bias[c];
        mx = fmaxf(mx, v[c]);
    }
    float s = 0.f;
    for (int c = 0; c < C; ++c) s += expf(v[c] - mx);
    float l = logf(s);
    for (int c = 0; c < C; ++c) out[(long long)n * C + c] = v[c] - mx - l;
}

// ---------------- host orchestration ----------------
static inline int cdiv(long long a, long long b) { return (int)((a + b - 1) / b); }

extern "C" void kernel_launch(void* const* d_in, const int* in_sizes, int n_in,
                              void* d_out, int out_size, void* d_ws, size_t ws_size,
                              hipStream_t stream) {
    const float* x    = (const float*)d_in[0];
    const int*   esrc = (const int*)d_in[1];
    const int    E    = in_sizes[1] / 2;
    const int*   edst = esrc + E;
    const int    N    = in_sizes[0] / 128;

    const float* W1 = (const float*)d_in[2];
    const float* as1 = (const float*)d_in[3];
    const float* ad1 = (const float*)d_in[4];
    const float* b1 = (const float*)d_in[5];
    const float* g1 = (const float*)d_in[6];
    const float* be1 = (const float*)d_in[7];
    const float* W2 = (const float*)d_in[8];
    const float* as2 = (const float*)d_in[9];
    const float* ad2 = (const float*)d_in[10];
    const float* b2 = (const float*)d_in[11];
    const float* g2 = (const float*)d_in[12];
    const float* be2 = (const float*)d_in[13];
    const float* W3 = (const float*)d_in[14];
    const float* as3 = (const float*)d_in[15];
    const float* ad3 = (const float*)d_in[16];
    const float* b3 = (const float*)d_in[17];

    // workspace layout (f32 region then bf16 region)
    size_t NF = (size_t)N * 256;
    float* h      = (float*)d_ws;
    float* xl     = h + NF;
    float* outacc = xl + NF;
    float* alsrc  = outacc + NF;
    float* aldst  = alsrc + (size_t)N * 8;
    float* amax   = aldst + (size_t)N * 8;
    float* denom  = amax + (size_t)N * 8;
    float* stats  = denom + (size_t)N * 8;
    unsigned short* hbf  = (unsigned short*)(stats + 1024);
    unsigned short* wtbf = hbf + NF;

    const int T = 256;
    auto edges = [&](int H, int C, const float* xlp, float* outp) {
        int NH = N * H;
        k_fill<<<cdiv(NH, T), T, 0, stream>>>(amax, -INFINITY, NH);
        k_fill<<<cdiv(NH, T), T, 0, stream>>>(denom, 0.0f, NH);
        k_fill<<<cdiv((long long)N * H * C, T), T, 0, stream>>>(outp, 0.0f, N * H * C);
        k_edge_max<<<cdiv((long long)E * H, T), T, 0, stream>>>(esrc, edst, alsrc, aldst, amax, E, H);
        k_fixmax<<<cdiv(NH, T), T, 0, stream>>>(amax, NH);
        k_edge_sum<<<cdiv((long long)E * H, T), T, 0, stream>>>(esrc, edst, alsrc, aldst, amax, denom, E, H);
        k_edge_scatter<<<cdiv((long long)E * H * C, T), T, 0, stream>>>(
            esrc, edst, alsrc, aldst, amax, denom, xlp, outp, E, H, C);
    };

    // ================= Layer 1: 128 -> 8x16 =================
    k_f32_to_bf16<<<cdiv((long long)N * 128, T), T, 0, stream>>>(x, hbf, N * 128);
    k_w_to_bf16_t<<<cdiv(128 * 128, T), T, 0, stream>>>(W1, wtbf, 128, 128, 128);
    {
        dim3 grid(cdiv(N, 16), 8);
        k_wmma_gemm_bf16<<<grid, 32, 0, stream>>>(hbf, wtbf, xl, N, 128, 128, 128);
    }
    k_scores<<<cdiv((long long)N * 8, T), T, 0, stream>>>(xl, as1, ad1, alsrc, aldst, N, 8, 16);
    edges(8, 16, xl, outacc);
    k_bn_stats<<<128, 256, 0, stream>>>(outacc, b1, stats, N, 128);
    k_bn_elu<<<cdiv((long long)N * 128, T), T, 0, stream>>>(outacc, b1, stats, g1, be1, h, N, 128);

    // ================= Layer 2: 128 -> 8x32 =================
    k_f32_to_bf16<<<cdiv((long long)N * 128, T), T, 0, stream>>>(h, hbf, N * 128);
    k_w_to_bf16_t<<<cdiv(256 * 128, T), T, 0, stream>>>(W2, wtbf, 128, 256, 256);
    {
        dim3 grid(cdiv(N, 16), 16);
        k_wmma_gemm_bf16<<<grid, 32, 0, stream>>>(hbf, wtbf, xl, N, 128, 256, 256);
    }
    k_scores<<<cdiv((long long)N * 8, T), T, 0, stream>>>(xl, as2, ad2, alsrc, aldst, N, 8, 32);
    edges(8, 32, xl, outacc);
    k_bn_stats<<<256, 256, 0, stream>>>(outacc, b2, stats, N, 256);
    k_bn_elu<<<cdiv((long long)N * 256, T), T, 0, stream>>>(outacc, b2, stats, g2, be2, h, N, 256);

    // ================= Layer 3: 256 -> 1x10 =================
    k_f32_to_bf16<<<cdiv((long long)N * 256, T), T, 0, stream>>>(h, hbf, N * 256);
    k_w_to_bf16_t<<<cdiv(16 * 256, T), T, 0, stream>>>(W3, wtbf, 256, 10, 16);
    {
        dim3 grid(cdiv(N, 16), 1);
        k_wmma_gemm_bf16<<<grid, 32, 0, stream>>>(hbf, wtbf, xl, N, 256, 10, 10);
    }
    k_scores<<<cdiv((long long)N * 1, T), T, 0, stream>>>(xl, as3, ad3, alsrc, aldst, N, 1, 10);
    edges(1, 10, xl, outacc);
    k_logsoftmax<<<cdiv(N, T), T, 0, stream>>>(outacc, b3, (float*)d_out, N, 10);
}
